// NodeCensNet_79791902425524
// MI455X (gfx1250) — matrix operand order repaired and checked by reading, not compiled, not run on hardware
//
#include <hip/hip_runtime.h>

// ---------------------------------------------------------------------------
// NodeCensNet fused pipeline for MI455X (gfx1250, wave32, WMMA bf16)
//
//   dinv[b,n]  = D[b,n,n] != 0 ? rsqrt(D[b,n,n]) : 0
//   HePe[b,e]  = edge[b,e,:] . edge_weight
//   S          = (T * HePe) @ T^T                      (GEMM1, bf16 WMMA)
//   M          = S * (adj + I) * dinv[col]             (mask epilogue)
//   out1       = M @ node                              (GEMM2, bf16 WMMA, fused)
//   out        = out1 @ node_weight                    (GEMM3, bf16 WMMA, fused)
//
// S and out1 never touch HBM. The full masked S tile is staged to LDS in one
// pass right after GEMM1 so acc1 dies early (avoids the round-1 VGPR spills);
// the S stage aliases the (dead) GEMM1 A/B tiles in shared memory.
// ---------------------------------------------------------------------------

typedef __attribute__((ext_vector_type(16))) __bf16        v16bf;
typedef __attribute__((ext_vector_type(8)))  float         v8f;
typedef __attribute__((ext_vector_type(4)))  float         f32x4;
typedef __attribute__((ext_vector_type(4)))  unsigned int  u32x4;
typedef __attribute__((ext_vector_type(2)))  unsigned int  u32x2;

constexpr int kB    = 4;
constexpr int kN    = 4096;
constexpr int kE    = 1024;
constexpr int kFIN  = 128;
constexpr int kFOUT = 128;
constexpr int kFE   = 64;
constexpr int TS    = 40;    // u16 row stride, 32-wide K tiles (32 + 8 pad)
constexpr int SS    = 136;   // u16 row stride, 128-wide S stage (128 + 8 pad)

static __device__ __forceinline__ unsigned short f2bf(float f) {
  return __builtin_bit_cast(unsigned short, (__bf16)f);
}
static __device__ __forceinline__ unsigned int pack2(float x, float y) {
  return (unsigned int)f2bf(x) | ((unsigned int)f2bf(y) << 16);
}

// A-matrix fragment gather (16x32 bf16, ISA 7.12.2): lane m = lane&15,
// half = lane>>4 selects K+8 inside each 16-K group. Two ds_load_b128.
static __device__ __forceinline__ v16bf load_frag_A(const unsigned short* lds,
                                                    int row, int stride,
                                                    int kbase, int lane) {
  const int m  = lane & 15;
  const int hf = lane >> 4;
  const unsigned short* p = lds + (row + m) * stride + kbase + hf * 8;
  union { u32x4 q[2]; v16bf v; } u;
  u.q[0] = *(const u32x4*)(p);        // K = kbase + hf*8 + 0..7
  u.q[1] = *(const u32x4*)(p + 16);   // K = kbase + 16 + hf*8 + 0..7
  return u.v;
}

// B-matrix fragment gather (32x16 bf16): lane n = lane&15, half selects K+8 inside each 16-K group.
// LDS holds B transposed: [n][k], k contiguous. Two ds_load_b128.
static __device__ __forceinline__ v16bf load_frag_B(const unsigned short* lds,
                                                    int col, int stride,
                                                    int kbase, int lane) {
  const int n  = lane & 15;
  const int hf = lane >> 4;
  const unsigned short* p = lds + (col + n) * stride + kbase + hf * 8;
  union { u32x4 q[2]; v16bf v; } u;
  u.q[0] = *(const u32x4*)(p);        // K = kbase + hf*8 + 0..7
  u.q[1] = *(const u32x4*)(p + 16);   // K = kbase + 16 + hf*8 + 0..7
  return u.v;
}

static __device__ __forceinline__ v8f wmma_bf16(v16bf a, v16bf b, v8f c) {
  // (neg_a, A, neg_b, B, c_mod, C, reuse_a, reuse_b)
  return __builtin_amdgcn_wmma_f32_16x16x32_bf16(false, a, false, b,
                                                 (short)0, c, false, false);
}

// ---------------------------------------------------------------------------
// Prep kernels
// ---------------------------------------------------------------------------
__global__ void prep_dinv_kernel(const float* __restrict__ D,
                                 float* __restrict__ dinv) {
  int i = blockIdx.x * blockDim.x + threadIdx.x;   // i = b*kN + n
  if (i >= kB * kN) return;
  int b = i >> 12;
  int n = i & (kN - 1);
  float d = D[(size_t)b * kN * kN + (size_t)n * (kN + 1)];
  dinv[i] = (d != 0.0f) ? rsqrtf(d) : 0.0f;
}

__global__ void prep_hepe_kernel(const float* __restrict__ edge,
                                 const float* __restrict__ ew,
                                 float* __restrict__ hepe) {
  int i = blockIdx.x * blockDim.x + threadIdx.x;   // i = b*kE + e
  if (i >= kB * kE) return;
  const float* ep = edge + (size_t)i * kFE;
  float s = 0.0f;
#pragma unroll
  for (int j = 0; j < kFE / 4; ++j) {
    f32x4 v = *(const f32x4*)(ep + 4 * j);
    f32x4 w = *(const f32x4*)(ew + 4 * j);
    s += v[0] * w[0] + v[1] * w[1] + v[2] * w[2] + v[3] * w[3];
  }
  hepe[i] = s;
}

// ---------------------------------------------------------------------------
// Fused main kernel: one workgroup = 128 output rows of one batch.
// 8 wave32s arranged 4 (row) x 2 (col); each wave owns 32x64 = 2x4 WMMA tiles.
// ---------------------------------------------------------------------------
__global__ void __launch_bounds__(256)
nodecensnet_fused_kernel(const float* __restrict__ node,
                         const float* __restrict__ adj,
                         const float* __restrict__ T,
                         const float* __restrict__ W,
                         const float* __restrict__ dinv,
                         const float* __restrict__ hepe,
                         float* __restrict__ out) {
  // smem layout:
  //   during GEMM1 K-loop : [ a_tile 128*TS | b_tile 128*TS ]  (20 KB)
  //   after GEMM1         : [ S stage 128*SS ]                 (34 KB, aliases)
  __shared__ __align__(16) unsigned short smem[128 * SS];
  __shared__ __align__(16) unsigned short c_lds[128 * TS];  // node/W chunk [n][k]

  unsigned short* a_lds = smem;                // Th tile  [row][k]
  unsigned short* b_lds = smem + 128 * TS;     // T^T tile [col][k]
  unsigned short* s_lds = smem;                // staged S / out1 (full 128x128)

  const int b  = blockIdx.y;
  const int n0 = blockIdx.x * 128;

  const int tid  = threadIdx.x;
  const int lane = tid & 31;
  const int wid  = tid >> 5;
  const int wr   = wid >> 1;   // 0..3 : row group (32 rows)
  const int wc   = wid & 1;    // 0..1 : col group (64 cols)
  const int lm   = lane & 15;
  const int hf   = lane >> 4;

  const float* Tb    = T    + (size_t)b * kN * kE;
  const float* adjb  = adj  + (size_t)b * kN * kN;
  const float* nodeb = node + (size_t)b * kN * kFIN;
  const float* dinvb = dinv + (size_t)b * kN;
  const float* hepeb = hepe + (size_t)b * kE;
  float*       outb  = out  + (size_t)b * kN * kFOUT;

  // Loader geometry for 128x32 tiles: each thread -> 4 rows x 4 cols (float4).
  const int lrow  = tid >> 3;          // 0..31
  const int lcol4 = (tid & 7) << 2;    // 0,4,...,28
  // Loader geometry for transposed 32(K)x128(F) chunks.
  const int cf4 = (tid & 31) << 2;     // f: 0..124
  const int ckb = tid >> 5;            // k: 0..7 (x4 iters)

  v8f acc2[8];                          // out1 accumulators (persist all m-blocks)
#pragma unroll
  for (int i = 0; i < 8; ++i) acc2[i] = {};

#pragma unroll 1
  for (int mb = 0; mb < kN / 128; ++mb) {
    const int m0 = mb * 128;

    v8f acc1[8];                        // S-tile accumulators
#pragma unroll
    for (int i = 0; i < 8; ++i) acc1[i] = {};

    // ---------------- GEMM1: S += Th[128 x 32] @ T^T[32 x 128] over E ------
    for (int e0 = 0; e0 < kE; e0 += 32) {
      {
        f32x4 hv = *(const f32x4*)(hepeb + e0 + lcol4);
#pragma unroll
        for (int i = 0; i < 4; ++i) {
          const int row = lrow + (i << 5);
          f32x4 av = *(const f32x4*)(Tb + (size_t)(n0 + row) * kE + e0 + lcol4);
          f32x4 bv = *(const f32x4*)(Tb + (size_t)(m0 + row) * kE + e0 + lcol4);
          u32x2 ap = { pack2(av[0] * hv[0], av[1] * hv[1]),
                       pack2(av[2] * hv[2], av[3] * hv[3]) };
          u32x2 bp = { pack2(bv[0], bv[1]), pack2(bv[2], bv[3]) };
          *(u32x2*)(&a_lds[row * TS + lcol4]) = ap;
          *(u32x2*)(&b_lds[row * TS + lcol4]) = bp;
        }
        if (e0 + 32 < kE) {   // emits global_prefetch_b8
          __builtin_prefetch(Tb + (size_t)(n0 + lrow) * kE + e0 + 32 + lcol4, 0, 1);
          __builtin_prefetch(Tb + (size_t)(m0 + lrow) * kE + e0 + 32 + lcol4, 0, 1);
        }
      }
      __syncthreads();

      v16bf af[2], bfv[4];
#pragma unroll
      for (int ti = 0; ti < 2; ++ti)
        af[ti] = load_frag_A(a_lds, (wr << 5) + (ti << 4), TS, 0, lane);
#pragma unroll
      for (int tj = 0; tj < 4; ++tj)
        bfv[tj] = load_frag_B(b_lds, (wc << 6) + (tj << 4), TS, 0, lane);
#pragma unroll
      for (int ti = 0; ti < 2; ++ti)
#pragma unroll
        for (int tj = 0; tj < 4; ++tj)
          acc1[ti * 4 + tj] = wmma_bf16(af[ti], bfv[tj], acc1[ti * 4 + tj]);
      __syncthreads();
    }

    // ------ mask epilogue: stage FULL masked S tile (kills acc1) -----------
    {
      float dv[4];
#pragma unroll
      for (int tj = 0; tj < 4; ++tj)
        dv[tj] = dinvb[m0 + (wc << 6) + (tj << 4) + lm];
#pragma unroll
      for (int ti = 0; ti < 2; ++ti)
#pragma unroll
        for (int tj = 0; tj < 4; ++tj) {
          const int a    = ti * 4 + tj;
          const int colL = (wc << 6) + (tj << 4) + lm;
          const int mg   = m0 + colL;
          const float dvv = dv[tj];
#pragma unroll
          for (int v = 0; v < 8; ++v) {
            const int rowL = (wr << 5) + (ti << 4) + v + (hf << 3);
            const int ng   = n0 + rowL;
            const float av = adjb[(size_t)ng * kN + mg];
            const float mask = (av + (ng == mg ? 1.0f : 0.0f)) * dvv;
            s_lds[rowL * SS + colL] = f2bf(acc1[a][v] * mask);
          }
        }
    }
    __syncthreads();

    // ------ GEMM2: out1 += M[128x128] @ node[128x128] (chunked K=32) -------
#pragma unroll 1
    for (int kc = 0; kc < 4; ++kc) {
      // c_lds[f][k] = bf16(node[m0 + kc*32 + k][f])  (transposed store)
#pragma unroll
      for (int i = 0; i < 4; ++i) {
        const int kk = ckb + (i << 3);
        f32x4 nv = *(const f32x4*)(nodeb + (size_t)(m0 + kc * 32 + kk) * kFIN + cf4);
#pragma unroll
        for (int c = 0; c < 4; ++c)
          c_lds[(cf4 + c) * TS + kk] = f2bf(nv[c]);
      }
      __syncthreads();

      v16bf af2[2], bf2[4];
#pragma unroll
      for (int ti = 0; ti < 2; ++ti)
        af2[ti] = load_frag_A(s_lds, (wr << 5) + (ti << 4), SS, kc * 32, lane);
#pragma unroll
      for (int tj = 0; tj < 4; ++tj)
        bf2[tj] = load_frag_B(c_lds, (wc << 6) + (tj << 4), TS, 0, lane);
#pragma unroll
      for (int ti = 0; ti < 2; ++ti)
#pragma unroll
        for (int tj = 0; tj < 4; ++tj)
          acc2[ti * 4 + tj] = wmma_bf16(af2[ti], bf2[tj], acc2[ti * 4 + tj]);
      __syncthreads();   // before next kc / next mb overwrites c_lds / smem
    }
  }

  // ---------------- GEMM3: out = out1 @ node_weight (chunked K=32) ---------
  v8f acc3[8];
#pragma unroll
  for (int i = 0; i < 8; ++i) acc3[i] = {};

  // Stage full out1 tile (kills acc2 before GEMM3 temps go live).
#pragma unroll
  for (int ti = 0; ti < 2; ++ti)
#pragma unroll
    for (int tj = 0; tj < 4; ++tj) {
      const int a    = ti * 4 + tj;
      const int colL = (wc << 6) + (tj << 4) + lm;
#pragma unroll
      for (int v = 0; v < 8; ++v) {
        const int rowL = (wr << 5) + (ti << 4) + v + (hf << 3);
        s_lds[rowL * SS + colL] = f2bf(acc2[a][v]);
      }
    }
  __syncthreads();

#pragma unroll 1
  for (int kc = 0; kc < 4; ++kc) {
    // c_lds[fout][k] = bf16(W[kc*32 + k][fout])  (transposed store)
#pragma unroll
    for (int i = 0; i < 4; ++i) {
      const int kk = ckb + (i << 3);
      f32x4 wv = *(const f32x4*)(W + (size_t)(kc * 32 + kk) * kFOUT + cf4);
#pragma unroll
      for (int c = 0; c < 4; ++c)
        c_lds[(cf4 + c) * TS + kk] = f2bf(wv[c]);
    }
    __syncthreads();

    v16bf af3[2], bf3[4];
#pragma unroll
    for (int ti = 0; ti < 2; ++ti)
      af3[ti] = load_frag_A(s_lds, (wr << 5) + (ti << 4), SS, kc * 32, lane);
#pragma unroll
    for (int tj = 0; tj < 4; ++tj)
      bf3[tj] = load_frag_B(c_lds, (wc << 6) + (tj << 4), TS, 0, lane);
#pragma unroll
    for (int ti = 0; ti < 2; ++ti)
#pragma unroll
      for (int tj = 0; tj < 4; ++tj)
        acc3[ti * 4 + tj] = wmma_bf16(af3[ti], bf3[tj], acc3[ti * 4 + tj]);
    __syncthreads();
  }

  // Write final f32 output.
#pragma unroll
  for (int ti = 0; ti < 2; ++ti)
#pragma unroll
    for (int tj = 0; tj < 4; ++tj) {
      const int a   = ti * 4 + tj;
      const int col = (wc << 6) + (tj << 4) + lm;
#pragma unroll
      for (int v = 0; v < 8; ++v) {
        const int rowL = (wr << 5) + (ti << 4) + v + (hf << 3);
        outb[(size_t)(n0 + rowL) * kFOUT + col] = acc3[a][v];
      }
    }
}

// ---------------------------------------------------------------------------
// Launch
// ---------------------------------------------------------------------------
extern "C" void kernel_launch(void* const* d_in, const int* in_sizes, int n_in,
                              void* d_out, int out_size, void* d_ws, size_t ws_size,
                              hipStream_t stream) {
  (void)in_sizes; (void)n_in; (void)out_size; (void)ws_size;
  const float* node   = (const float*)d_in[0];   // [B,N,F_IN]
  const float* edge   = (const float*)d_in[1];   // [B,E,F_EDGE]
  const float* adj    = (const float*)d_in[2];   // [B,N,N]
  const float* D      = (const float*)d_in[3];   // [B,N,N]
  const float* T      = (const float*)d_in[4];   // [B,N,E]
  const float* ew     = (const float*)d_in[5];   // [F_EDGE]
  const float* W      = (const float*)d_in[6];   // [F_IN,F_OUT]
  float*       out    = (float*)d_out;           // [B,N,F_OUT]

  float* ws   = (float*)d_ws;
  float* dinv = ws;                 // B*N floats
  float* hepe = ws + kB * kN;       // B*E floats

  prep_dinv_kernel<<<(kB * kN + 255) / 256, 256, 0, stream>>>(D, dinv);
  prep_hepe_kernel<<<(kB * kE + 255) / 256, 256, 0, stream>>>(edge, ew, hepe);

  dim3 grid(kN / 128, kB);
  nodecensnet_fused_kernel<<<grid, 256, 0, stream>>>(node, adj, T, W,
                                                     dinv, hepe, out);
}
